// TransformerModule_81578608820635
// MI455X (gfx1250) — matrix operand order repaired and checked by reading, not compiled, not run on hardware
//
#include <hip/hip_runtime.h>
#include <hip/hip_bf16.h>
#include <math.h>

// ---------------------------------------------------------------------------
// Fused equivariant edge-attention for MI455X (gfx1250), wave32 + WMMA.
// One wave32 per destination node (its 16 inbound edges are contiguous).
// All per-edge MLP GEMMs run on v_wmma_f32_16x16x32_f16 with M=16 edges.
// ---------------------------------------------------------------------------

typedef __attribute__((ext_vector_type(16))) _Float16 v16h;
typedef __attribute__((ext_vector_type(8)))  _Float16 v8h;
typedef __attribute__((ext_vector_type(8)))  float    v8f;
typedef __attribute__((ext_vector_type(4)))  float    v4f;

#define NNODES 8192
#define KEDGE  16
#define DFEAT  64
#define SSH    8
#define BINV   64
#define HIDDIM 64
#define FDIM   576      // D*(1+S)
#define NT3    36       // FDIM/16
#define WAVES  2        // waves (nodes) per block

// Prepacked weight fragments in d_ws (f16). One frag = 32 lanes x 16 halves.
#define FRAG_ELEMS 512
#define WK1_OFF 0                    // 4 nt x 2 kt = 8 frags
#define WK2_OFF (8*FRAG_ELEMS)       // 8 frags
#define WK3_OFF (16*FRAG_ELEMS)      // 36 nt x 2 kt = 72 frags
#define WV1_OFF (88*FRAG_ELEMS)
#define WV2_OFF (96*FRAG_ELEMS)
#define WV3_OFF (104*FRAG_ELEMS)
#define NFRAGS  176                  // 176*512*2B = 180224 bytes of d_ws

#define SYNC_DS() asm volatile("s_wait_dscnt 0x0" ::: "memory")

struct __align__(16) WaveLds {
    float    xs[16][64];    // x_src per edge
    float    q[64];         // node_feat[dst]
    float    sh[16][8];     // edge_sh
    float    cut[16];       // edge_weight_cutoff
    float    scal[16][128]; // [:, :64] = q*k0 ; [:, 64+d] = raw sum_s sh^2*out2k
    float    logit[16][4];
    float    salpha[16][4]; // sqrt(relu(alpha))
    float    nodeacc[576];  // per-node segment_sum of edge_v
    _Float16 h[16][64];     // hidden activations (f16, A-fragment source)
};

// Branch-free tanh-approximate GELU:
//   gelu(x) = 0.5*x*(1 + tanh(c*(x + 0.044715 x^3))) = x * sigmoid(2c*(x + ...))
// sigmoid computed with the hw transcendentals (v_exp_f32 + v_rcp_f32),
// avoiding libm tanhf's exec-mask branch sequences in the hot epilogue.
__device__ __forceinline__ float gelu_tanh(float x) {
    const float k = 1.5957691216057308f;  // 2 * sqrt(2/pi)
    float y = k * (x + 0.044715f * x * x * x);
    float e = __expf(-y);                 // v_exp_f32 (branchless)
    return x * __builtin_amdgcn_rcpf(1.0f + e);
}

// A-fragment (16x32 f16, MxK) from a per-lane f32 row pointer.
// Lane layout (ISA 7.12.2): lane L row m=L&15; halves 0..7 -> K=kb..kb+7,
// halves 8..15 -> K=kb+16..kb+23, with kb = (L>=16 ? 8 : 0).
__device__ __forceinline__ v16h a_from_f32(const float* p) {
    v16h a;
#pragma unroll
    for (int i = 0; i < 8; ++i) a[i]     = (_Float16)p[i];
#pragma unroll
    for (int i = 0; i < 8; ++i) a[8 + i] = (_Float16)p[16 + i];
    return a;
}

__device__ __forceinline__ v16h a_from_lds(const _Float16* p) {
    v8h lo = *(const v8h*)(p);
    v8h hi = *(const v8h*)(p + 16);
    v16h a;
#pragma unroll
    for (int i = 0; i < 8; ++i) { a[i] = lo[i]; a[8 + i] = hi[i]; }
    return a;
}

__device__ __forceinline__ v16h b_frag(const _Float16* ws, int lane) {
    return *(const v16h*)(ws + lane * 16);
}

__device__ __forceinline__ v8f wmma_f16(v16h a, v16h b, v8f c) {
    return __builtin_amdgcn_wmma_f32_16x16x32_f16(false, a, false, b,
                                                  (short)0, c, false, false);
}

// Layers 1+2 of the edge MLP (16 edges x 64 -> 64 -> 64, gelu after each).
// Result (f16) left in L.h.
__device__ __forceinline__ void mlp12(WaveLds& L, const float* inv_row,
                                      const _Float16* w1, const _Float16* w2,
                                      int lane, int kb) {
    const int mrow = (lane >> 4) * 8;   // C/D layout row base
    const int n    = lane & 15;
    // ---- layer 1: A straight from global edge_inv (f32 -> f16) ----
    v16h a0 = a_from_f32(inv_row + 0  + kb);
    v16h a1 = a_from_f32(inv_row + 32 + kb);
#pragma unroll
    for (int nt = 0; nt < 4; ++nt) {
        v8f c = {};
        c = wmma_f16(a0, b_frag(w1 + (nt * 2 + 0) * FRAG_ELEMS, lane), c);
        c = wmma_f16(a1, b_frag(w1 + (nt * 2 + 1) * FRAG_ELEMS, lane), c);
#pragma unroll
        for (int r = 0; r < 8; ++r)
            L.h[mrow + r][nt * 16 + n] = (_Float16)gelu_tanh(c[r]);
    }
    SYNC_DS();
    // ---- layer 2: A from LDS h (DS ops are in-order within the wave) ----
    v16h A0 = a_from_lds(&L.h[lane & 15][0  + kb]);
    v16h A1 = a_from_lds(&L.h[lane & 15][32 + kb]);
    v8f cc[4];
#pragma unroll
    for (int nt = 0; nt < 4; ++nt) {
        v8f c = {};
        c = wmma_f16(A0, b_frag(w2 + (nt * 2 + 0) * FRAG_ELEMS, lane), c);
        c = wmma_f16(A1, b_frag(w2 + (nt * 2 + 1) * FRAG_ELEMS, lane), c);
        cc[nt] = c;
    }
#pragma unroll
    for (int nt = 0; nt < 4; ++nt)
#pragma unroll
        for (int r = 0; r < 8; ++r)
            L.h[mrow + r][nt * 16 + n] = (_Float16)gelu_tanh(cc[nt][r]);
    SYNC_DS();
}

// ---------------------------------------------------------------------------
// Kernel 1: pack weight matrices into WMMA B-fragment layout (f16) in d_ws.
// Frag for (nt,kt) of W[Kdim x Ndim]: lane l supplies column n = nt*16+(l&15);
// half<8: K = kt*32 + kb + half; half>=8: K = kt*32 + 16 + kb + (half-8), kb = (l>=16 ? 8 : 0).
// ---------------------------------------------------------------------------
__global__ void pack_weights(const float* __restrict__ Wk1, const float* __restrict__ Wk2,
                             const float* __restrict__ Wk3, const float* __restrict__ Wv1,
                             const float* __restrict__ Wv2, const float* __restrict__ Wv3,
                             _Float16* __restrict__ ws) {
    const int frag = blockIdx.x;           // 0..175
    const int lane = threadIdx.x >> 4;     // 0..31
    const int half = threadIdx.x & 15;     // 0..15
    const float* W; int Ndim, local;
    if      (frag < 8)   { W = Wk1; Ndim = 64;  local = frag; }
    else if (frag < 16)  { W = Wk2; Ndim = 64;  local = frag - 8; }
    else if (frag < 88)  { W = Wk3; Ndim = 576; local = frag - 16; }
    else if (frag < 96)  { W = Wv1; Ndim = 64;  local = frag - 88; }
    else if (frag < 104) { W = Wv2; Ndim = 64;  local = frag - 96; }
    else                 { W = Wv3; Ndim = 576; local = frag - 104; }
    const int nt = local >> 1, kt = local & 1;
    const int k  = kt * 32 + ((lane < 16) ? 0 : 8) + (half & 7) + ((half >> 3) * 16);
    const int n  = nt * 16 + (lane & 15);
    ws[frag * FRAG_ELEMS + threadIdx.x] = (_Float16)W[k * Ndim + n];
}

// ---------------------------------------------------------------------------
// Kernel 2: one wave32 per node — fully fused attention block.
// ---------------------------------------------------------------------------
__global__ void __launch_bounds__(WAVES * 32)
edge_attn(const float* __restrict__ node_feat, const float* __restrict__ edge_sh,
          const float* __restrict__ edge_inv,  const float* __restrict__ cutoff,
          const float* __restrict__ W_logit,   const float* __restrict__ W_out,
          const int*   __restrict__ edge_src,  const _Float16* __restrict__ wfrag,
          float* __restrict__ out) {
    __shared__ WaveLds lds[WAVES];
    const int wave = threadIdx.x >> 5;
    const int lane = threadIdx.x & 31;
    const int node = blockIdx.x * WAVES + wave;
    WaveLds& L = lds[wave];
    const int e0   = node * KEDGE;
    const int m16  = lane & 15;
    const int kb   = (lane >> 4) * 8;
    const int mrow = (lane >> 4) * 8;
    const int n    = lane & 15;

    // ---- stage node/edge data into LDS ----
    L.q[lane]      = node_feat[node * DFEAT + lane];
    L.q[32 + lane] = node_feat[node * DFEAT + 32 + lane];
    if (lane < 16) {
        L.cut[lane] = cutoff[e0 + lane];
#pragma unroll
        for (int i = 0; i < SSH; ++i) L.sh[lane][i] = edge_sh[(e0 + lane) * SSH + i];
    }
    {
        const int src  = edge_src[e0 + m16];
        const int base = (lane >> 4) * 32;
        const float* sp = node_feat + (size_t)src * DFEAT + base;
#pragma unroll
        for (int i = 0; i < 32; i += 4)
            *(v4f*)&L.xs[m16][base + i] = *(const v4f*)(sp + i);
    }
    {
        float* sp = &L.scal[0][0];
#pragma unroll
        for (int i = 0; i < 64; ++i) sp[i * 32 + lane] = 0.0f;
    }
    __builtin_amdgcn_wave_barrier();
    SYNC_DS();

    const float* invrow = edge_inv + (size_t)(e0 + m16) * BINV;

    // ================= KEY path =================
    mlp12(L, invrow, wfrag + WK1_OFF, wfrag + WK2_OFF, lane, kb);
    {
        v16h A0 = a_from_lds(&L.h[m16][0  + kb]);
        v16h A1 = a_from_lds(&L.h[m16][32 + kb]);
        for (int nt = 0; nt < NT3; ++nt) {
            // prefetch B fragments a few tiles ahead (global_prefetch_b8)
            if (nt + 4 < NT3)
                __builtin_prefetch(wfrag + WK3_OFF + ((nt + 4) * 2) * FRAG_ELEMS + lane * 16, 0, 3);
            v8f c = {};
            c = wmma_f16(A0, b_frag(wfrag + WK3_OFF + (nt * 2 + 0) * FRAG_ELEMS, lane), c);
            c = wmma_f16(A1, b_frag(wfrag + WK3_OFF + (nt * 2 + 1) * FRAG_ELEMS, lane), c);
            const int cidx = nt * 16 + n;
            if (nt < 4) {            // edge_key[:, :64] -> scal0 = q*x_src*out2k
#pragma unroll
                for (int r = 0; r < 8; ++r) {
                    const int m = mrow + r;
                    L.scal[m][cidx] = c[r] * L.xs[m][cidx] * L.q[cidx];
                }
            } else {                 // scal1 raw: sum_s sh^2 * out2k (d = f>>3)
                const int f = cidx - 64, d = f >> 3, s = f & 7;
#pragma unroll
                for (int r = 0; r < 8; ++r) {
                    const int m = mrow + r;
                    const float sv = L.sh[m][s];
                    atomicAdd(&L.scal[m][64 + d], c[r] * sv * sv);
                }
            }
        }
    }
    SYNC_DS();

    // ---- logits: scal(16x128) @ W_logit(128x4), one edge per lane 0..15 ----
    if (lane < 16) {
        float lg0 = 0, lg1 = 0, lg2 = 0, lg3 = 0;
        const int m = lane;
#pragma unroll 4
        for (int d = 0; d < 64; ++d) {
            const float sv = L.scal[m][d];
            lg0 += sv * W_logit[d * 4 + 0]; lg1 += sv * W_logit[d * 4 + 1];
            lg2 += sv * W_logit[d * 4 + 2]; lg3 += sv * W_logit[d * 4 + 3];
        }
#pragma unroll 4
        for (int d = 0; d < 64; ++d) {   // apply deferred q*x_src factor
            const float sv = L.scal[m][64 + d] * L.q[d] * L.xs[m][d];
            lg0 += sv * W_logit[(64 + d) * 4 + 0]; lg1 += sv * W_logit[(64 + d) * 4 + 1];
            lg2 += sv * W_logit[(64 + d) * 4 + 2]; lg3 += sv * W_logit[(64 + d) * 4 + 3];
        }
        L.logit[m][0] = lg0; L.logit[m][1] = lg1;
        L.logit[m][2] = lg2; L.logit[m][3] = lg3;
    }
    SYNC_DS();

    // ---- segmented softmax over the node's 16 edges (one head per lane) ----
    if (lane < 4) {
        float mx = -INFINITY;
        for (int m = 0; m < 16; ++m) mx = fmaxf(mx, L.logit[m][lane]);
        float ex[16], z = 0.f;
        for (int m = 0; m < 16; ++m) {
            ex[m] = L.cut[m] * __expf(L.logit[m][lane] - mx);
            z += ex[m];
        }
        if (z == 0.f) z = 1.f;
        const float rz = __builtin_amdgcn_rcpf(z);
        for (int m = 0; m < 16; ++m)
            L.salpha[m][lane] = sqrtf(fmaxf(ex[m] * rz, 0.f));
    }
    SYNC_DS();

    // ================= VALUE path =================
    mlp12(L, invrow, wfrag + WV1_OFF, wfrag + WV2_OFF, lane, kb);
    {
        v16h A0 = a_from_lds(&L.h[m16][0  + kb]);
        v16h A1 = a_from_lds(&L.h[m16][32 + kb]);
        for (int nt = 0; nt < NT3; ++nt) {
            if (nt + 4 < NT3)
                __builtin_prefetch(wfrag + WV3_OFF + ((nt + 4) * 2) * FRAG_ELEMS + lane * 16, 0, 3);
            v8f c = {};
            c = wmma_f16(A0, b_frag(wfrag + WV3_OFF + (nt * 2 + 0) * FRAG_ELEMS, lane), c);
            c = wmma_f16(A1, b_frag(wfrag + WV3_OFF + (nt * 2 + 1) * FRAG_ELEMS, lane), c);
            const int cidx = nt * 16 + n;
            const int head = nt / 9;     // 144 cols per head, tile-aligned
            float colsum = 0.f;
#pragma unroll
            for (int r = 0; r < 8; ++r) {
                const int m = mrow + r;
                float o1;
                if (cidx < 64) o1 = L.xs[m][cidx];
                else { const int f = cidx - 64; o1 = L.xs[m][f >> 3] * L.sh[m][f & 7]; }
                colsum += c[r] * o1 * L.salpha[m][head];
            }
            colsum += __shfl_xor(colsum, 16);  // fold rows 8..15 onto 0..7
            if (lane < 16) L.nodeacc[cidx] = colsum;
        }
    }
    SYNC_DS();

    // ---- final: node_out(1x576) @ W_out(576x64); 2 output cols per lane ----
    {
        const float2* w2 = (const float2*)W_out;  // row stride 32 float2
        float acc0 = 0.f, acc1 = 0.f;
#pragma unroll 4
        for (int f = 0; f < FDIM; ++f) {
            const float a = L.nodeacc[f];
            const float2 w = w2[f * 32 + lane];
            acc0 += a * w.x;
            acc1 += a * w.y;
        }
        out[node * DFEAT + lane * 2]     = acc0;
        out[node * DFEAT + lane * 2 + 1] = acc1;
    }
}

// ---------------------------------------------------------------------------
extern "C" void kernel_launch(void* const* d_in, const int* in_sizes, int n_in,
                              void* d_out, int out_size, void* d_ws, size_t ws_size,
                              hipStream_t stream) {
    const float* node_feat = (const float*)d_in[0];
    const float* edge_sh   = (const float*)d_in[1];
    const float* edge_inv  = (const float*)d_in[2];
    const float* cutoff    = (const float*)d_in[3];
    const float* Wk1       = (const float*)d_in[4];
    const float* Wk2       = (const float*)d_in[5];
    const float* Wk3       = (const float*)d_in[6];
    const float* Wv1       = (const float*)d_in[7];
    const float* Wv2       = (const float*)d_in[8];
    const float* Wv3       = (const float*)d_in[9];
    const float* W_logit   = (const float*)d_in[10];
    const float* W_out     = (const float*)d_in[11];
    const int*   edge_src  = (const int*)d_in[12];
    (void)in_sizes; (void)n_in; (void)out_size;

    _Float16* wfrag = (_Float16*)d_ws;   // needs NFRAGS*512*2 = 180224 B
    (void)ws_size;

    pack_weights<<<NFRAGS, 512, 0, stream>>>(Wk1, Wk2, Wk3, Wv1, Wv2, Wv3, wfrag);
    edge_attn<<<NNODES / WAVES, WAVES * 32, 0, stream>>>(
        node_feat, edge_sh, edge_inv, cutoff, W_logit, W_out, edge_src,
        wfrag, (float*)d_out);
}
